// MultiInResidualBlock_66030827209068
// MI455X (gfx1250) — matrix operand order repaired and checked by reading, not compiled, not run on hardware
//
#include <hip/hip_runtime.h>
#include <math.h>

// ---------------------------------------------------------------------------
// MultiInResidualBlock for MI455X (gfx1250).
// GEMMs use bf16x3 split-precision on V_WMMA_F32_16X16X32_BF16:
//   a*b ~= ah*bh + ah*bl + al*bh   (limbs capture 16 mantissa bits, ~1e-5 rel)
// N=8192 tokens, D=256, E=4 experts, H=1024, O=768, top-2 routing.
// ---------------------------------------------------------------------------

#define N_TOK 8192
#define DDIM  256
#define NEXP  4
#define HDIM  1024
#define ODIM  768
#define TOPK  2
#define LN_EPS 1e-6f

typedef float  v4f  __attribute__((ext_vector_type(4)));
typedef float  v8f  __attribute__((ext_vector_type(8)));
typedef __bf16 v8bf  __attribute__((ext_vector_type(8)));
typedef __bf16 v16bf __attribute__((ext_vector_type(16)));
typedef unsigned short u16;

#define XSTR 264    // bf16 row stride in kernel 1 LDS (264*2B=528B, 4 mod 64 DW)
#define HSTR 1032   // bf16 row stride in kernel 2 LDS (2064B, 4 mod 64 DW)

__device__ __forceinline__ v8f vzero8() {
    v8f z = {0.f,0.f,0.f,0.f,0.f,0.f,0.f,0.f};
    return z;
}
__device__ __forceinline__ unsigned bf16_hi_bits(float f) {
    return __builtin_bit_cast(unsigned, f) >> 16;   // truncating bf16
}
__device__ __forceinline__ float from_hi_bits(unsigned h) {
    return __builtin_bit_cast(float, h << 16);
}
// split f into two bf16 limbs (bit patterns)
__device__ __forceinline__ void bf16_split(float f, unsigned& h, unsigned& l) {
    h = bf16_hi_bits(f);
    l = bf16_hi_bits(f - from_hi_bits(h));          // exact residual, truncated
}

// D(16x16,f32) += A(16x32,bf16) * B(32x16,bf16)
// A: lane l holds K-chunks [8*(l>>4), +8) and [16+8*(l>>4), +8) of row l&15
// B: lane l holds B[k][l&15] for K-chunks [8*(l>>4), +8) and [16+8*(l>>4), +8)  (== W[col][k])
// C/D: lane l, vgpr r -> element [r + 8*(l>>4)][l&15]
__device__ __forceinline__ v8f wmma_bf16(v16bf a, v16bf b, v8f c) {
    return __builtin_amdgcn_wmma_f32_16x16x32_bf16(
        /*neg_a=*/false, a, /*neg_b=*/false, b,
        /*c_mod=*/(short)0, c, /*reuse_a=*/false, /*reuse_b=*/false);
}

__device__ __forceinline__ v16bf cat8(v8bf a, v8bf b) {
    return __builtin_shufflevector(a, b, 0,1,2,3,4,5,6,7,8,9,10,11,12,13,14,15);
}

// ---------------------------------------------------------------------------
// Prep: elementwise split fp32 -> bf16 hi/lo limb buffers (for weights).
// ---------------------------------------------------------------------------
__global__ __launch_bounds__(256)
void split_bf16_kernel(const float* __restrict__ src,
                       u16* __restrict__ hi, u16* __restrict__ lo, int n4)
{
    for (int i = blockIdx.x * blockDim.x + threadIdx.x; i < n4;
         i += gridDim.x * blockDim.x) {
        v4f v = ((const v4f*)src)[i];
        unsigned h[4], l[4];
        #pragma unroll
        for (int j = 0; j < 4; ++j) bf16_split(v[j], h[j], l[j]);
        uint2 ph = { h[0] | (h[1] << 16), h[2] | (h[3] << 16) };
        uint2 pl = { l[0] | (l[1] << 16), l[2] | (l[3] << 16) };
        ((uint2*)hi)[i] = ph;
        ((uint2*)lo)[i] = pl;
    }
}

// ---------------------------------------------------------------------------
// Kernel 1: dense-over-experts MoE GEMM + combine (+ optional exact GELU).
// One block = 16 tokens; 8 waves sweep OUTD in 128-column passes.
// SPLIT_OUT=true  -> write bf16 hi/lo limbs (hid path, consumed by kernel 2)
// SPLIT_OUT=false -> write fp32 (res path)
// ---------------------------------------------------------------------------
template<int OUTD, bool GELU, bool SPLIT_OUT>
__global__ __launch_bounds__(256)
void moe_gemm_kernel(const float* __restrict__ xf,
                     const float* __restrict__ ew,
                     const int*   __restrict__ ei,
                     const u16*   __restrict__ Whi,
                     const u16*   __restrict__ Wlo,
                     const float* __restrict__ bvec,
                     u16* __restrict__ out_hi,
                     u16* __restrict__ out_lo,
                     float* __restrict__ out_f)
{
    __shared__ u16 xs_hi[NEXP * 16 * XSTR];
    __shared__ u16 xs_lo[NEXP * 16 * XSTR];
    __shared__ float sc[16][NEXP];

    const int tid  = threadIdx.x;
    const int wave = tid >> 5;
    const int lane = tid & 31;
    const int hf   = lane >> 4;   // lane half
    const int lo   = lane & 15;
    const int m0   = blockIdx.x * 16;

    // ---- per-token expert scales s[n,e] (top-2, weight-normalized) ----
    if (tid < 16) {
        const int n = m0 + tid;
        const float w0 = ew[n * TOPK + 0], w1 = ew[n * TOPK + 1];
        const int   i0 = ei[n * TOPK + 0], i1 = ei[n * TOPK + 1];
        const bool  v0 = (unsigned)i0 < (unsigned)NEXP;
        const bool  v1 = (unsigned)i1 < (unsigned)NEXP;
        float rs = (v0 ? w0 : 0.f) + (v1 ? w1 : 0.f);
        if (rs == 0.f) rs = 1.f;
        float s[NEXP] = {0.f, 0.f, 0.f, 0.f};
        if (v0) s[i0] += w0 / rs;
        if (v1) s[i1] += w1 / rs;
        #pragma unroll
        for (int e = 0; e < NEXP; ++e) sc[tid][e] = s[e];
    }

    // ---- stage X for all 4 experts as bf16 limbs (nan->0, expert mask) ----
    for (int it = tid; it < NEXP * 16 * (DDIM / 4); it += 256) {
        const int e = it >> 10;         // / (16 * 64)
        const int r = (it >> 6) & 15;   // token row in tile
        const int d = (it & 63) * 4;    // D offset
        v4f v = *(const v4f*)(xf + ((size_t)(e * N_TOK + m0 + r)) * DDIM + d);
        const int lim = 32 * (e + 1);   // IN_DIMS[e]
        unsigned h[4], l[4];
        #pragma unroll
        for (int j = 0; j < 4; ++j) {
            float f = v[j];
            if (f != f) f = 0.f;                  // nan_to_num
            if (((d + j) & 127) >= lim) f = 0.f;  // mask both halves
            bf16_split(f, h[j], l[j]);
        }
        uint2 ph = { h[0] | (h[1] << 16), h[2] | (h[3] << 16) };
        uint2 pl = { l[0] | (l[1] << 16), l[2] | (l[3] << 16) };
        *(uint2*)(&xs_hi[(e * 16 + r) * XSTR + d]) = ph;
        *(uint2*)(&xs_lo[(e * 16 + r) * XSTR + d]) = pl;
    }
    __syncthreads();

    // ---- bf16x3 WMMA passes over output columns ----
    for (int col0 = wave * 16; col0 < OUTD; col0 += 128) {
        const int o = col0 + lo;
        v8f acc = vzero8();
        #pragma unroll
        for (int e = 0; e < NEXP; ++e) {
            v8f pe = vzero8();
            const u16* xah = &xs_hi[(e * 16 + lo) * XSTR + 8 * hf];
            const u16* xal = &xs_lo[(e * 16 + lo) * XSTR + 8 * hf];
            const u16* wbh = Whi + ((size_t)e * OUTD + o) * DDIM + 8 * hf;
            const u16* wbl = Wlo + ((size_t)e * OUTD + o) * DDIM + 8 * hf;
            #pragma unroll
            for (int kc = 0; kc < DDIM; kc += 32) {
                v16bf Ah = cat8(*(const v8bf*)(xah + kc),
                                *(const v8bf*)(xah + kc + 16));
                v16bf Al = cat8(*(const v8bf*)(xal + kc),
                                *(const v8bf*)(xal + kc + 16));
                v16bf Bh = cat8(*(const v8bf*)(wbh + kc),
                                *(const v8bf*)(wbh + kc + 16));
                v16bf Bl = cat8(*(const v8bf*)(wbl + kc),
                                *(const v8bf*)(wbl + kc + 16));
                pe = wmma_bf16(Ah, Bh, pe);
                pe = wmma_bf16(Ah, Bl, pe);
                pe = wmma_bf16(Al, Bh, pe);
            }
            #pragma unroll
            for (int r = 0; r < 8; ++r)
                pe[r] *= sc[r + 8 * hf][e], acc[r] += pe[r];  // row-wise scale
        }
        const float b0 = bvec[0 * OUTD + o];
        const float b1 = bvec[1 * OUTD + o];
        const float b2 = bvec[2 * OUTD + o];
        const float b3 = bvec[3 * OUTD + o];
        #pragma unroll
        for (int r = 0; r < 8; ++r) {
            const int M = r + 8 * hf;
            float val = acc[r] + sc[M][0] * b0 + sc[M][1] * b1
                               + sc[M][2] * b2 + sc[M][3] * b3;
            if (GELU)
                val = 0.5f * val * (1.f + erff(val * 0.70710678118654752f));
            const size_t idx = (size_t)(m0 + M) * OUTD + o;
            if (SPLIT_OUT) {
                unsigned h, l;
                bf16_split(val, h, l);
                out_hi[idx] = (u16)h;
                out_lo[idx] = (u16)l;
            } else {
                out_f[idx] = val;
            }
        }
    }
}

// ---------------------------------------------------------------------------
// Kernel 2: out = hid @ W_out^T + b_out + res, fused RMS-norm, write d_out.
// One block = 16 tokens; 8 waves x 6 tiles = 768 columns. bf16x3 GEMM.
// ---------------------------------------------------------------------------
__global__ __launch_bounds__(256)
void out_ln_kernel(const u16* __restrict__ hid_hi,
                   const u16* __restrict__ hid_lo,
                   const u16* __restrict__ Whi,
                   const u16* __restrict__ Wlo,
                   const float* __restrict__ bout,
                   const float* __restrict__ res,
                   const float* __restrict__ lnw,
                   float* __restrict__ out)
{
    __shared__ u16 hs_hi[16 * HSTR];
    __shared__ u16 hs_lo[16 * HSTR];
    __shared__ float ssq[16];
    __shared__ float rsf[16];

    const int tid  = threadIdx.x;
    const int wave = tid >> 5;
    const int lane = tid & 31;
    const int hf   = lane >> 4;
    const int lo   = lane & 15;
    const int m0   = blockIdx.x * 16;

    if (tid < 16) ssq[tid] = 0.f;

    // stage hid limb tiles (16 x 1024 bf16 each), uint4 = 8 bf16
    for (int it = tid; it < 16 * (HDIM / 8); it += 256) {
        const int r = it >> 7;
        const int d = (it & 127) * 8;
        *(uint4*)(&hs_hi[r * HSTR + d]) =
            *(const uint4*)(hid_hi + (size_t)(m0 + r) * HDIM + d);
        *(uint4*)(&hs_lo[r * HSTR + d]) =
            *(const uint4*)(hid_lo + (size_t)(m0 + r) * HDIM + d);
    }
    __syncthreads();

    float vals[6][8];
    float ps[8];
    #pragma unroll
    for (int r = 0; r < 8; ++r) ps[r] = 0.f;

    #pragma unroll
    for (int t = 0; t < 6; ++t) {
        const int col0 = (wave * 6 + t) * 16;
        const int o = col0 + lo;
        v8f a8 = vzero8();
        const u16* xah = &hs_hi[lo * HSTR + 8 * hf];
        const u16* xal = &hs_lo[lo * HSTR + 8 * hf];
        const u16* wbh = Whi + (size_t)o * HDIM + 8 * hf;
        const u16* wbl = Wlo + (size_t)o * HDIM + 8 * hf;
        #pragma unroll 4
        for (int kc = 0; kc < HDIM; kc += 32) {
            v16bf Ah = cat8(*(const v8bf*)(xah + kc),
                            *(const v8bf*)(xah + kc + 16));
            v16bf Al = cat8(*(const v8bf*)(xal + kc),
                            *(const v8bf*)(xal + kc + 16));
            v16bf Bh = cat8(*(const v8bf*)(wbh + kc),
                            *(const v8bf*)(wbh + kc + 16));
            v16bf Bl = cat8(*(const v8bf*)(wbl + kc),
                            *(const v8bf*)(wbl + kc + 16));
            a8 = wmma_bf16(Ah, Bh, a8);
            a8 = wmma_bf16(Ah, Bl, a8);
            a8 = wmma_bf16(Al, Bh, a8);
        }
        const float bo = bout[o];
        #pragma unroll
        for (int r = 0; r < 8; ++r) {
            const int n = m0 + r + 8 * hf;
            float val = a8[r] + bo + res[(size_t)n * ODIM + o];
            vals[t][r] = val;
            ps[r] += val * val;
        }
    }

    // row-wise sum of squares: butterfly over the 16 lanes sharing a token
    #pragma unroll
    for (int r = 0; r < 8; ++r) {
        float p = ps[r];
        p += __shfl_xor(p, 1);
        p += __shfl_xor(p, 2);
        p += __shfl_xor(p, 4);
        p += __shfl_xor(p, 8);
        if (lo == 0) atomicAdd(&ssq[r + 8 * hf], p);
    }
    __syncthreads();
    if (tid < 16) rsf[tid] = rsqrtf(ssq[tid] / (float)ODIM + LN_EPS);
    __syncthreads();

    #pragma unroll
    for (int t = 0; t < 6; ++t) {
        const int o = (wave * 6 + t) * 16 + lo;
        const float lw = lnw[o];
        #pragma unroll
        for (int r = 0; r < 8; ++r) {
            const int M = r + 8 * hf;
            out[(size_t)(m0 + M) * ODIM + o] = lw * vals[t][r] * rsf[M];
        }
    }
}

// ---------------------------------------------------------------------------
extern "C" void kernel_launch(void* const* d_in, const int* in_sizes, int n_in,
                              void* d_out, int out_size, void* d_ws, size_t ws_size,
                              hipStream_t stream)
{
    // setup_inputs order: x, in_feat_size, expert_weights, expert_indices,
    // x_final, W_hid, b_hid, W_out, b_out, W_res, b_res, ln_w
    const float* ew   = (const float*)d_in[2];
    const int*   ei   = (const int*)  d_in[3];
    const float* xf   = (const float*)d_in[4];
    const float* Whid = (const float*)d_in[5];
    const float* bhid = (const float*)d_in[6];
    const float* Wout = (const float*)d_in[7];
    const float* bo   = (const float*)d_in[8];
    const float* Wres = (const float*)d_in[9];
    const float* bres = (const float*)d_in[10];
    const float* lnw  = (const float*)d_in[11];
    float* outp = (float*)d_out;

    // workspace layout (bytes)
    char* p = (char*)d_ws;
    u16*   hid_hi  = (u16*)p;   p += (size_t)N_TOK * HDIM * 2;        // 16 MB
    u16*   hid_lo  = (u16*)p;   p += (size_t)N_TOK * HDIM * 2;        // 16 MB
    float* res_ws  = (float*)p; p += (size_t)N_TOK * ODIM * 4;        // 24 MB
    u16*   Whid_hi = (u16*)p;   p += (size_t)NEXP * HDIM * DDIM * 2;  //  2 MB
    u16*   Whid_lo = (u16*)p;   p += (size_t)NEXP * HDIM * DDIM * 2;
    u16*   Wout_hi = (u16*)p;   p += (size_t)ODIM * HDIM * 2;         // 1.5 MB
    u16*   Wout_lo = (u16*)p;   p += (size_t)ODIM * HDIM * 2;
    u16*   Wres_hi = (u16*)p;   p += (size_t)NEXP * ODIM * DDIM * 2;  // 1.5 MB
    u16*   Wres_lo = (u16*)p;   p += (size_t)NEXP * ODIM * DDIM * 2;

    dim3 blk(256);
    // weight limb splits (cheap, once per call)
    split_bf16_kernel<<<512, blk, 0, stream>>>(Whid, Whid_hi, Whid_lo,
                                               NEXP * HDIM * DDIM / 4);
    split_bf16_kernel<<<512, blk, 0, stream>>>(Wout, Wout_hi, Wout_lo,
                                               ODIM * HDIM / 4);
    split_bf16_kernel<<<512, blk, 0, stream>>>(Wres, Wres_hi, Wres_lo,
                                               NEXP * ODIM * DDIM / 4);

    dim3 grid(N_TOK / 16);
    moe_gemm_kernel<HDIM, true,  true ><<<grid, blk, 0, stream>>>(
        xf, ew, ei, Whid_hi, Whid_lo, bhid, hid_hi, hid_lo, nullptr);
    moe_gemm_kernel<ODIM, false, false><<<grid, blk, 0, stream>>>(
        xf, ew, ei, Wres_hi, Wres_lo, bres, nullptr, nullptr, res_ws);
    out_ln_kernel<<<grid, blk, 0, stream>>>(
        hid_hi, hid_lo, Wout_hi, Wout_lo, bo, res_ws, lnw, outp);
}
